// AnswerModule_40776419508833
// MI455X (gfx1250) — matrix operand order repaired and checked by reading, not compiled, not run on hardware
//
#include <hip/hip_runtime.h>
#include <hip/hip_bf16.h>

// Problem constants (match reference)
#define B_SZ   64
#define H_SZ   512
#define P_LEN  512
#define Q_LEN  64
#define T_STEPS 4
#define D_SZ   1024
#define NEG_BIG (-1e9f)

typedef __attribute__((ext_vector_type(16))) __bf16 v16bf;
typedef __attribute__((ext_vector_type(2)))  __bf16 v2bf;
typedef __attribute__((ext_vector_type(8)))  float  v8f;

__device__ __forceinline__ float bf2f(unsigned short s) {
  unsigned int u = ((unsigned int)s) << 16;
  return __builtin_bit_cast(float, u);
}
// Pack two fp32 -> two bf16 in one dword (single v_cvt_pk_bf16_f32 if present)
__device__ __forceinline__ unsigned int pack2bf(float a, float b) {
#if __has_builtin(__builtin_amdgcn_cvt_pk_bf16_f32)
  v2bf v = __builtin_amdgcn_cvt_pk_bf16_f32(a, b);
  return __builtin_bit_cast(unsigned int, v);
#else
  __bf16 x = (__bf16)a, y = (__bf16)b;
  unsigned short ux = __builtin_bit_cast(unsigned short, x);
  unsigned short uy = __builtin_bit_cast(unsigned short, y);
  return ((unsigned int)uy << 16) | (unsigned int)ux;
#endif
}

// ---------------------------------------------------------------------------
// Generic batched GEMM: C[b][m][n] = sum_k A[b][m][k] * B[k][n], K = 1024.
// All strides are template constants -> pure immediate/scalar addressing.
// fp32 sources converted to bf16 on the fly; f32 accumulate via
// v_wmma_f32_16x16x32_bf16. WG = 128 threads (4 waves), tile 64(M)x64(N),
// K stepped by 64 (8 WMMAs per barrier pair). Dims divide tile sizes.
// ---------------------------------------------------------------------------
template<int LDA_R, int LDA_C, int BATCH_A,
         int LDB_R, int LDB_C,
         int LDC, int BATCH_C, int OUT_BF16>
__global__ __launch_bounds__(128)
void gemm_bf16_wmma(const float* __restrict__ A,
                    const float* __restrict__ Bm,
                    void* __restrict__ C)
{
  __shared__ __align__(16) unsigned short ldsA[64][72]; // [m][k], padded rows
  __shared__ __align__(16) unsigned short ldsB[64][72]; // [n][k], padded rows

  const int tid  = threadIdx.x;
  const int lane = tid & 31;
  const int wv   = tid >> 5;
  const int l15  = lane & 15;
  const int half = lane >> 4;
  const int m0 = blockIdx.y * 64;
  const int n0 = blockIdx.x * 64;
  const float* Ab = A + (long long)blockIdx.z * BATCH_A;

  v8f acc[4];
  #pragma unroll
  for (int j = 0; j < 4; ++j)
    #pragma unroll
    for (int i = 0; i < 8; ++i) acc[j][i] = 0.0f;

  for (int k0 = 0; k0 < D_SZ; k0 += 64) {
    // A tile 64m x 64k = 2048 bf16-pairs, 16 pairs/thread, one b32 DS store
    #pragma unroll
    for (int i = 0; i < 16; ++i) {
      int idx = tid + i * 128;
      int mm, kp;
      if constexpr (LDA_R == 1) { mm = idx & 63; kp = idx >> 6; }
      else                      { kp = idx & 31; mm = idx >> 5; }
      const float* p = Ab + (long long)(m0 + mm) * LDA_R
                          + (long long)(k0 + 2 * kp) * LDA_C;
      *(unsigned int*)&ldsA[mm][2 * kp] = pack2bf(p[0], p[LDA_C]);
    }
    // B tile 64n x 64k
    #pragma unroll
    for (int i = 0; i < 16; ++i) {
      int idx = tid + i * 128;
      int nn, kp;
      if constexpr (LDB_C == 1) { nn = idx & 63; kp = idx >> 6; }
      else                      { kp = idx & 31; nn = idx >> 5; }
      const float* p = Bm + (long long)(k0 + 2 * kp) * LDB_R
                          + (long long)(n0 + nn) * LDB_C;
      *(unsigned int*)&ldsB[nn][2 * kp] = pack2bf(p[0], p[LDB_R]);
    }
    __syncthreads();

    #pragma unroll
    for (int ks = 0; ks < 2; ++ks) {           // two 32-wide K slabs
      // A fragment (16x32 bf16), ISA layout: half 0 -> K {0..7,16..23},
      // half 1 -> K {8..15,24..31}; two contiguous 16B LDS runs.
      v16bf afrag;
      {
        const __bf16* r0 = (const __bf16*)&ldsA[wv * 16 + l15][ks * 32 + half * 8];
        const __bf16* r1 = (const __bf16*)&ldsA[wv * 16 + l15][ks * 32 + half * 8 + 16];
        #pragma unroll
        for (int e = 0; e < 8; ++e) { afrag[e] = r0[e]; afrag[8 + e] = r1[e]; }
      }
      #pragma unroll
      for (int j = 0; j < 4; ++j) {
        // B fragment (32x16): half 0 -> K {0..7,16..23}, half 1 -> K {8..15,24..31}
        v16bf bfrag;
        const __bf16* rb = (const __bf16*)&ldsB[j * 16 + l15][ks * 32 + half * 8];
        #pragma unroll
        for (int e = 0; e < 16; ++e) bfrag[e] = rb[(e < 8) ? e : e + 8];
        acc[j] = __builtin_amdgcn_wmma_f32_16x16x32_bf16(
            false, afrag, false, bfrag, (short)0, acc[j], false, false);
      }
    }
    __syncthreads();
  }

  // Epilogue: C/D layout — lane l15 = n, VGPR i -> m = half*8 + i
  if constexpr (OUT_BF16) {
    __bf16* Cb = (__bf16*)C + (long long)blockIdx.z * BATCH_C;
    #pragma unroll
    for (int j = 0; j < 4; ++j) {
      int n = n0 + j * 16 + l15;
      #pragma unroll
      for (int i = 0; i < 8; ++i) {
        int m = m0 + wv * 16 + half * 8 + i;
        Cb[(long long)m * LDC + n] = (__bf16)acc[j][i];
      }
    }
  } else {
    float* Cb = (float*)C + (long long)blockIdx.z * BATCH_C;
    #pragma unroll
    for (int j = 0; j < 4; ++j) {
      int n = n0 + j * 16 + l15;
      #pragma unroll
      for (int i = 0; i < 8; ++i) {
        int m = m0 + wv * 16 + half * 8 + i;
        Cb[(long long)m * LDC + n] = acc[j][i];
      }
    }
  }
}

// ---------------------------------------------------------------------------
// Zero-fill fp32 buffer
// ---------------------------------------------------------------------------
__global__ void zero_f32(float* __restrict__ p, int n) {
  int i = blockIdx.x * blockDim.x + threadIdx.x;
  if (i < n) p[i] = 0.0f;
}

// ---------------------------------------------------------------------------
// Question attention, fused: sq = proj_q(bf16) . v5 + qm ; alpha = softmax_q ;
// c[b,d] = sum_q H_q[b,d,q] * alpha[q].  One WG of 256 per batch.
// ---------------------------------------------------------------------------
__global__ __launch_bounds__(256)
void attn_ctx(const unsigned short* __restrict__ projQ,
              const float* __restrict__ v5,
              const unsigned char* __restrict__ qmask,
              const float* __restrict__ H_q,
              float* __restrict__ cOut)
{
  __shared__ __align__(16) float vsh[H_SZ];
  __shared__ float part[256];
  __shared__ float sq[Q_LEN];
  __shared__ __align__(16) float alpha[Q_LEN];
  __shared__ float sc[2];

  const int b = blockIdx.x, tid = threadIdx.x;
  vsh[tid]       = v5[(size_t)b * H_SZ + tid];
  vsh[tid + 256] = v5[(size_t)b * H_SZ + tid + 256];
  __syncthreads();

  const int q = tid & 63, seg = tid >> 6;
  {
    const unsigned short* row =
        projQ + ((size_t)b * Q_LEN + q) * H_SZ + seg * 128;
    float acc = 0.0f;
    #pragma unroll
    for (int h = 0; h < 128; h += 8) {
      uint4 pk = *(const uint4*)(row + h);
      const float* vp = &vsh[seg * 128 + h];
      acc += bf2f((unsigned short)(pk.x & 0xFFFF)) * vp[0];
      acc += bf2f((unsigned short)(pk.x >> 16))    * vp[1];
      acc += bf2f((unsigned short)(pk.y & 0xFFFF)) * vp[2];
      acc += bf2f((unsigned short)(pk.y >> 16))    * vp[3];
      acc += bf2f((unsigned short)(pk.z & 0xFFFF)) * vp[4];
      acc += bf2f((unsigned short)(pk.z >> 16))    * vp[5];
      acc += bf2f((unsigned short)(pk.w & 0xFFFF)) * vp[6];
      acc += bf2f((unsigned short)(pk.w >> 16))    * vp[7];
    }
    part[tid] = acc;
  }
  __syncthreads();
  if (tid < Q_LEN) {
    float s = part[tid] + part[64 + tid] + part[128 + tid] + part[192 + tid];
    s += qmask[(size_t)b * Q_LEN + tid] ? 0.0f : NEG_BIG;
    sq[tid] = s;
  }
  __syncthreads();
  if (tid == 0) {
    float mx = sq[0];
    for (int i = 1; i < Q_LEN; ++i) mx = fmaxf(mx, sq[i]);
    sc[0] = mx;
  }
  __syncthreads();
  if (tid < Q_LEN) alpha[tid] = __expf(sq[tid] - sc[0]);
  __syncthreads();
  if (tid == 0) {
    float sum = 0.0f;
    for (int i = 0; i < Q_LEN; ++i) sum += alpha[i];
    sc[1] = 1.0f / sum;
  }
  __syncthreads();
  if (tid < Q_LEN) alpha[tid] *= sc[1];
  __syncthreads();

  // c[b,d] = sum_q H_q[b,d,q] * alpha[q]
  #pragma unroll
  for (int i = 0; i < 4; ++i) {
    int d = tid + i * 256;
    const float4* hq = (const float4*)(H_q + ((size_t)b * D_SZ + d) * Q_LEN);
    float acc = 0.0f;
    #pragma unroll
    for (int q4 = 0; q4 < 16; ++q4) {
      float4 hv = hq[q4];
      float4 av = *(const float4*)&alpha[q4 * 4];
      acc += hv.x * av.x + hv.y * av.y + hv.z * av.z + hv.w * av.w;
    }
    cOut[(size_t)b * D_SZ + d] = acc;
  }
}

// ---------------------------------------------------------------------------
// Passage scores, fused: s[p] = proj_p(bf16) . v7 + pm ; softmax_p ; write to
// one or two output slices (p2[t] == p1[t+1]). One WG of 256 per batch.
// ---------------------------------------------------------------------------
__global__ __launch_bounds__(256)
void score_softmax_p(const unsigned short* __restrict__ projP,
                     const float* __restrict__ v7,
                     const unsigned char* __restrict__ pmask,
                     float* __restrict__ dst1, float* __restrict__ dst2)
{
  __shared__ __align__(16) float vsh[H_SZ];
  __shared__ float s[P_LEN];
  __shared__ float red[256];

  const int b = blockIdx.x, tid = threadIdx.x;
  vsh[tid]       = v7[(size_t)b * H_SZ + tid];
  vsh[tid + 256] = v7[(size_t)b * H_SZ + tid + 256];
  __syncthreads();

  float sv[2];
  #pragma unroll
  for (int r = 0; r < 2; ++r) {
    int p = tid + r * 256;
    const unsigned short* row = projP + ((size_t)b * P_LEN + p) * H_SZ;
    float acc = 0.0f;
    for (int h = 0; h < H_SZ; h += 8) {
      uint4 pk = *(const uint4*)(row + h);
      acc += bf2f((unsigned short)(pk.x & 0xFFFF)) * vsh[h + 0];
      acc += bf2f((unsigned short)(pk.x >> 16))    * vsh[h + 1];
      acc += bf2f((unsigned short)(pk.y & 0xFFFF)) * vsh[h + 2];
      acc += bf2f((unsigned short)(pk.y >> 16))    * vsh[h + 3];
      acc += bf2f((unsigned short)(pk.z & 0xFFFF)) * vsh[h + 4];
      acc += bf2f((unsigned short)(pk.z >> 16))    * vsh[h + 5];
      acc += bf2f((unsigned short)(pk.w & 0xFFFF)) * vsh[h + 6];
      acc += bf2f((unsigned short)(pk.w >> 16))    * vsh[h + 7];
    }
    acc += pmask[(size_t)b * P_LEN + p] ? 0.0f : NEG_BIG;
    s[p] = acc; sv[r] = acc;
  }
  __syncthreads();

  red[tid] = fmaxf(s[tid], s[tid + 256]);
  __syncthreads();
  for (int st = 128; st > 0; st >>= 1) {
    if (tid < st) red[tid] = fmaxf(red[tid], red[tid + st]);
    __syncthreads();
  }
  float mx = red[0];
  __syncthreads();

  float e0 = __expf(sv[0] - mx);
  float e1 = __expf(sv[1] - mx);
  red[tid] = e0 + e1;
  __syncthreads();
  for (int st = 128; st > 0; st >>= 1) {
    if (tid < st) red[tid] += red[tid + st];
    __syncthreads();
  }
  float inv = 1.0f / red[0];

  size_t o = (size_t)b * P_LEN + tid;
  float o0 = e0 * inv, o1 = e1 * inv;
  dst1[o] = o0; dst1[o + 256] = o1;
  if (dst2) { dst2[o] = o0; dst2[o + 256] = o1; }
}

// ---------------------------------------------------------------------------
// GRU cell elementwise (PyTorch GRUCell math)
// ---------------------------------------------------------------------------
__global__ __launch_bounds__(256)
void gru_cell(const float* __restrict__ gi, const float* __restrict__ gh,
              const float* __restrict__ bih, const float* __restrict__ bhh,
              const float* __restrict__ h, float* __restrict__ hNew)
{
  int idx = blockIdx.x * blockDim.x + threadIdx.x;   // 64*1024
  int b = idx >> 10, d = idx & 1023;
  size_t base = (size_t)b * (3 * D_SZ);
  float i_r = gi[base + d]            + bih[d];
  float i_z = gi[base + D_SZ + d]     + bih[D_SZ + d];
  float i_n = gi[base + 2 * D_SZ + d] + bih[2 * D_SZ + d];
  float h_r = gh[base + d]            + bhh[d];
  float h_z = gh[base + D_SZ + d]     + bhh[D_SZ + d];
  float h_n = gh[base + 2 * D_SZ + d] + bhh[2 * D_SZ + d];
  float r = 1.0f / (1.0f + __expf(-(i_r + h_r)));
  float z = 1.0f / (1.0f + __expf(-(i_z + h_z)));
  float n = tanhf(i_n + r * h_n);
  size_t ho = (size_t)b * D_SZ + d;
  hNew[ho] = (1.0f - z) * n + z * h[ho];
}

// ---------------------------------------------------------------------------
extern "C" void kernel_launch(void* const* d_in, const int* in_sizes, int n_in,
                              void* d_out, int out_size, void* d_ws, size_t ws_size,
                              hipStream_t stream)
{
  (void)in_sizes; (void)n_in; (void)out_size; (void)ws_size;
  // setup_inputs order: H_p(unused values), H_q, M, W_4..W_7, gru_W_ih,
  // gru_W_hh, gru_b_ih, gru_b_hh, p_mask, q_mask
  const float* H_q = (const float*)d_in[1];
  const float* M   = (const float*)d_in[2];
  const float* W4  = (const float*)d_in[3];
  const float* W5  = (const float*)d_in[4];
  const float* W6  = (const float*)d_in[5];
  const float* W7  = (const float*)d_in[6];
  const float* Wih = (const float*)d_in[7];
  const float* Whh = (const float*)d_in[8];
  const float* bih = (const float*)d_in[9];
  const float* bhh = (const float*)d_in[10];
  const unsigned char* pmask = (const unsigned char*)d_in[11];
  const unsigned char* qmask = (const unsigned char*)d_in[12];
  float* out = (float*)d_out;

  // Workspace carve-up (~40 MB)
  char* ws = (char*)d_ws;
  unsigned short* projP = (unsigned short*)ws; ws += (size_t)B_SZ * P_LEN * H_SZ * 2;
  unsigned short* projQ = (unsigned short*)ws; ws += (size_t)B_SZ * Q_LEN * H_SZ * 2;
  float* h0 = (float*)ws; ws += (size_t)B_SZ * D_SZ * 4;
  float* h1 = (float*)ws; ws += (size_t)B_SZ * D_SZ * 4;
  float* v5 = (float*)ws; ws += (size_t)B_SZ * H_SZ * 4;
  float* v7 = (float*)ws; ws += (size_t)B_SZ * H_SZ * 4;
  float* cb = (float*)ws; ws += (size_t)B_SZ * D_SZ * 4;
  float* gi = (float*)ws; ws += (size_t)B_SZ * 3 * D_SZ * 4;
  float* gh = (float*)ws; ws += (size_t)B_SZ * 3 * D_SZ * 4;

  const dim3 blk128(128), blk256(256);

  // Template instantiations:
  //  projP : A[m=p,k=d] = M[b,d,p]      -> LDA_R=1,    LDA_C=P_LEN, bf16 out
  //  projQ : A[m=q,k=d] = H_q[b,d,q]    -> LDA_R=1,    LDA_C=Q_LEN, bf16 out
  //  state : A[m=b,k=d] = h[b,d]        -> LDA_R=D_SZ, LDA_C=1,  B=W[k,n]
  //  gru   : B[k,n] = W[n,k] (W^T)      -> LDB_R=1,    LDB_C=D_SZ, N=3D
  auto gemm_projP = gemm_bf16_wmma<1, P_LEN, D_SZ * P_LEN,
                                   H_SZ, 1, H_SZ, P_LEN * H_SZ, 1>;
  auto gemm_projQ = gemm_bf16_wmma<1, Q_LEN, D_SZ * Q_LEN,
                                   H_SZ, 1, H_SZ, Q_LEN * H_SZ, 1>;
  auto gemm_state = gemm_bf16_wmma<D_SZ, 1, 0, H_SZ, 1, H_SZ, 0, 0>;
  auto gemm_gru   = gemm_bf16_wmma<D_SZ, 1, 0, 1, D_SZ, 3 * D_SZ, 0, 0>;

  // h0 = 0
  zero_f32<<<(B_SZ * D_SZ + 255) / 256, blk256, 0, stream>>>(h0, B_SZ * D_SZ);

  // proj_p[b,p,h] = sum_d M[b,d,p] * W6[d,h]  (bf16 out)
  gemm_projP<<<dim3(H_SZ / 64, P_LEN / 64, B_SZ), blk128, 0, stream>>>(M, W6, projP);

  // proj_q[b,q,h] = sum_d H_q[b,d,q] * W4[d,h]  (bf16 out)
  gemm_projQ<<<dim3(H_SZ / 64, Q_LEN / 64, B_SZ), blk128, 0, stream>>>(H_q, W4, projQ);

  // s_0 with h_0 = 0:  v7 = h0 @ W7 ; p1[0] = softmax(projP . v7 + pm)
  gemm_state<<<dim3(H_SZ / 64, B_SZ / 64, 1), blk128, 0, stream>>>(h0, W7, v7);
  score_softmax_p<<<B_SZ, blk256, 0, stream>>>(
      projP, v7, pmask, out /* p1[0] */, nullptr);

  const size_t SLICE = (size_t)B_SZ * P_LEN;   // one [B,P] slice
  for (int t = 0; t < T_STEPS; ++t) {
    float* hA = (t & 1) ? h1 : h0;   // h_t
    float* hB = (t & 1) ? h0 : h1;   // h_{t+1}

    // v5 = h @ W5
    gemm_state<<<dim3(H_SZ / 64, B_SZ / 64, 1), blk128, 0, stream>>>(hA, W5, v5);

    // alpha = softmax(proj_q . v5 + qm) ; c = H_q @ alpha
    attn_ctx<<<B_SZ, blk256, 0, stream>>>(projQ, v5, qmask, H_q, cb);

    // gi = c @ Wih^T ; gh = h @ Whh^T   (N = 3*D)
    gemm_gru<<<dim3(3 * D_SZ / 64, B_SZ / 64, 1), blk128, 0, stream>>>(cb, Wih, gi);
    gemm_gru<<<dim3(3 * D_SZ / 64, B_SZ / 64, 1), blk128, 0, stream>>>(hA, Whh, gh);

    // h_{t+1} = GRU(c, h_t)
    gru_cell<<<(B_SZ * D_SZ) / 256, blk256, 0, stream>>>(gi, gh, bih, bhh, hA, hB);

    // s_{t+1}: v7 = h_{t+1} @ W7 ; softmax -> p2[t] (and p1[t+1] if t<T-1)
    gemm_state<<<dim3(H_SZ / 64, B_SZ / 64, 1), blk128, 0, stream>>>(hB, W7, v7);
    float* dst1 = out + (size_t)(T_STEPS + t) * SLICE;          // p2[t]
    float* dst2 = (t < T_STEPS - 1) ? out + (size_t)(t + 1) * SLICE : nullptr;
    score_softmax_p<<<B_SZ, blk256, 0, stream>>>(projP, v7, pmask, dst1, dst2);
  }
}
